// Informer_19052474925600
// MI455X (gfx1250) — matrix operand order repaired and checked by reading, not compiled, not run on hardware
//
#include <hip/hip_runtime.h>
#include <hip/hip_bf16.h>
#include <math.h>

// ---------------- problem constants (Informer reference) ----------------
constexpr int kB   = 8;
constexpr int kL   = 2048;
constexpr int kC   = 16;
constexpr int kD   = 512;     // d_model
constexpr int kH   = 8;       // heads
constexpr int kDh  = 64;      // head dim
constexpr int kF   = 2048;    // d_ff
constexpr int kT   = 4;       // time features
constexpr int kU   = 24;      // FACTOR * ceil(log(L)) = 3*8
constexpr int kBL  = kB * kL; // 16384 rows

typedef __bf16 bf16_t;
typedef __attribute__((ext_vector_type(16))) __bf16 v16bf;
typedef __attribute__((ext_vector_type(8)))  float  v8f;
typedef unsigned int u32x4 __attribute__((ext_vector_type(4)));
typedef int          i32x8 __attribute__((ext_vector_type(8)));
typedef int          i32x4 __attribute__((ext_vector_type(4)));

#if defined(__HIP_DEVICE_COMPILE__) && __has_builtin(__builtin_amdgcn_tensor_load_to_lds) && __has_builtin(__builtin_amdgcn_s_wait_tensorcnt)
#define USE_TDM 1
#else
#define USE_TDM 0
#endif

// ---------------------------------------------------------------------------
// Pack fp32 weight (K x N row-major) into WMMA B-fragment order, bf16.
// Tile (nt,kt) holds 512 bf16: element lane*16+j  <->  B[kt*32 + 16*(lane>>4) + j,
//                                                        nt*16 + (lane&15)]
// Tiles stored nt-major, so a block's 64-column B panel is 4 tile-rows with
// global row stride ktiles*1KB -> a single 2-D TDM descriptor per K-step.
// ---------------------------------------------------------------------------
__global__ __launch_bounds__(256) void pack_w_kernel(
    const float* __restrict__ W, bf16_t* __restrict__ Wp, int K, int N)
{
    int idx = blockIdx.x * 256 + threadIdx.x;
    if (idx >= K * N) return;
    int tile = idx >> 9;
    int e    = idx & 511;
    int lane = e >> 4, j = e & 15;
    int half = lane >> 4, lr = lane & 15;
    int ktiles = K >> 5;
    int nt = tile / ktiles, kt = tile - nt * ktiles;
    int kk  = kt * 32 + half * 16 + j;
    int col = nt * 16 + lr;
    Wp[idx] = (bf16_t)W[(size_t)kk * N + col];
}

__global__ __launch_bounds__(256) void f32_to_bf16_kernel(
    const float* __restrict__ s, bf16_t* __restrict__ d, int n)
{
    int i = blockIdx.x * 256 + threadIdx.x;
    if (i < n) d[i] = (bf16_t)s[i];
}

// ---------------------------------------------------------------------------
// bf16 WMMA GEMM: C[M,N] = act(A[M,K] @ B[K,N] + bias)
//  - block = 256 threads = 8 wave32, output tile 128(M) x 64(N)
//  - wave w: rows [bx*128+w*16, +16), 4 accumulators across 64 cols
//  - B panel (4 tiles x 1KB per K-step) staged into LDS by the Tensor Data
//    Mover (double-buffered, s_wait_tensorcnt + barriers); fragments read
//    back with ds_load_b128.  A fragments: contiguous global b128 loads.
//  act: 0=none, 1=exact GELU, 2=ReLU.  Writes fp32 (Cf) and/or bf16 (Cb).
// ---------------------------------------------------------------------------
__global__ __launch_bounds__(256) void gemm_bf16_wmma(
    const bf16_t* __restrict__ A, const bf16_t* __restrict__ Bp,
    const float* __restrict__ bias, float* __restrict__ Cf,
    bf16_t* __restrict__ Cb, int M, int N, int K, int act)
{
    __shared__ __align__(16) bf16_t bsh[2][2048];   // 2 x 4KB B-panel chunks
    const int tid  = threadIdx.x;
    const int lane = tid & 31;
    const int wave = tid >> 5;
    const int half = lane >> 4, lr = lane & 15;
    const int mRow = (blockIdx.x * 8 + wave) * 16;
    const int nt0  = blockIdx.y * 4;                // 4 N-tiles = 64 columns
    const int ktiles = K >> 5;

    const bf16_t* ap     = A  + (size_t)(mRow + lr) * K + half * 16;
    const bf16_t* bpanel = Bp + (size_t)nt0 * ktiles * 512;

    // escape the LDS buffer so the compiler keeps the ds loads (TDM writes it)
    { void* p = (void*)&bsh[0][0]; asm volatile("" :: "v"(p) : "memory"); }

#if USE_TDM
    const unsigned lds0 = (unsigned)(size_t)(&bsh[0][0]);
    // D# group1: data_size=8B; 2-D tile 128x4; global row stride ktiles*128
    i32x8 g1;
    g1[0] = 3 << 16;          // [17:16] data_size = 8B
    g1[1] = 128 << 16;        // [79:48] tensor_dim0 = 128 (lo16)
    g1[2] = 4 << 16;          // dim0 hi = 0, [111:80] tensor_dim1 = 4 (lo16)
    g1[3] = 128 << 16;        // dim1 hi = 0, [127:112] tile_dim0 = 128
    g1[4] = 4;                // [143:128] tile_dim1 = 4, tile_dim2 = 0
    g1[5] = ktiles * 128;     // [207:160] tensor_dim0_stride (8B units)
    g1[6] = 0;
    g1[7] = 0;
    const i32x4 z4 = {0, 0, 0, 0};
    const i32x8 z8 = {0, 0, 0, 0, 0, 0, 0, 0};
#endif

    v8f acc[4];
#pragma unroll
    for (int t = 0; t < 4; ++t)
        acc[t] = (v8f){0.f, 0.f, 0.f, 0.f, 0.f, 0.f, 0.f, 0.f};

#if USE_TDM
    if (wave == 0) {
        unsigned long long ga = (unsigned long long)(size_t)bpanel;
        u32x4 g0;
        g0[0] = 1u;                                   // count = 1 descriptor
        g0[1] = lds0;                                 // lds_addr (buf 0)
        g0[2] = (unsigned)(ga & 0xffffffffull);
        g0[3] = (unsigned)((ga >> 32) & 0x01fffffful) | (2u << 30); // type=2
        __builtin_amdgcn_tensor_load_to_lds(g0, g1, z4, z4, z8, 0);
    }
#endif

    for (int kt = 0; kt < ktiles; ++kt) {
#if USE_TDM
        const int buf = kt & 1;
        if (wave == 0) {
            if (kt + 1 < ktiles) {
                unsigned long long ga =
                    (unsigned long long)(size_t)(bpanel + (size_t)(kt + 1) * 512);
                u32x4 g0;
                g0[0] = 1u;
                g0[1] = lds0 + (unsigned)((buf ^ 1) * 4096);
                g0[2] = (unsigned)(ga & 0xffffffffull);
                g0[3] = (unsigned)((ga >> 32) & 0x01fffffful) | (2u << 30);
                __builtin_amdgcn_tensor_load_to_lds(g0, g1, z4, z4, z8, 0);
                __builtin_amdgcn_s_wait_tensorcnt(1);   // current chunk done
            } else {
                __builtin_amdgcn_s_wait_tensorcnt(0);
            }
        }
        __syncthreads();                                // chunk visible to all
#else
        const int buf = 0;
        {   // cooperative staging: 4KB = 256 threads x 16B
            int t = tid >> 6, q = tid & 63;
            const bf16_t* s = bpanel + ((size_t)t * ktiles + kt) * 512 + q * 8;
            bf16_t* dst = &bsh[0][t * 512 + q * 8];
#pragma unroll
            for (int j = 0; j < 8; ++j) dst[j] = s[j];
        }
        __syncthreads();
#endif
        // A fragment (rows of this wave), contiguous -> 2x global_load_b128
        v16bf a;
#pragma unroll
        for (int j = 0; j < 8; ++j) { a[j] = ap[j]; a[j + 8] = ap[8 + j]; }
        __builtin_prefetch(ap + 32, 0, 1);

#pragma unroll
        for (int t = 0; t < 4; ++t) {
            const bf16_t* bfrag = &bsh[buf][t * 512 + lane * 16];
            v16bf b;
#pragma unroll
            for (int j = 0; j < 16; ++j) b[j] = bfrag[j];
            acc[t] = __builtin_amdgcn_wmma_f32_16x16x32_bf16(
                false, a, false, b, (short)0, acc[t], false, false);
        }
        __syncthreads();      // reads done before this buffer is refilled
        ap += 32;
    }

    // epilogue: bias + activation, fp32 and/or bf16 store
#pragma unroll
    for (int t = 0; t < 4; ++t) {
        const int col = nt0 * 16 + t * 16 + lr;
        const float bv = bias ? bias[col] : 0.f;
#pragma unroll
        for (int r = 0; r < 8; ++r) {
            float vv = acc[t][r] + bv;
            if (act == 1)      vv = 0.5f * vv * (1.f + erff(vv * 0.70710678f));
            else if (act == 2) vv = fmaxf(vv, 0.f);
            size_t off = (size_t)(mRow + half * 8 + r) * N + col;
            if (Cf) Cf[off] = vv;
            if (Cb) Cb[off] = (bf16_t)vv;
        }
    }
    (void)M;
}

// ---------------------------------------------------------------------------
// LayerNorm over D=512, optional fused pre-add: out = LN(x [+ add]) * g + b
// ---------------------------------------------------------------------------
__global__ __launch_bounds__(256) void ln512_kernel(
    const float* __restrict__ x, const float* __restrict__ add,
    const float* __restrict__ g, const float* __restrict__ bt,
    float* __restrict__ out)
{
    const int row = blockIdx.x, tid = threadIdx.x;
    const size_t base = (size_t)row * kD;
    float v0 = x[base + tid], v1 = x[base + tid + 256];
    if (add) { v0 += add[base + tid]; v1 += add[base + tid + 256]; }
    __shared__ float s[256];
    s[tid] = v0 + v1;
    __syncthreads();
    for (int o = 128; o > 0; o >>= 1) { if (tid < o) s[tid] += s[tid + o]; __syncthreads(); }
    float mean = s[0] * (1.f / kD);
    __syncthreads();
    float d0 = v0 - mean, d1 = v1 - mean;
    s[tid] = d0 * d0 + d1 * d1;
    __syncthreads();
    for (int o = 128; o > 0; o >>= 1) { if (tid < o) s[tid] += s[tid + o]; __syncthreads(); }
    float inv = rsqrtf(s[0] * (1.f / kD) + 1e-5f);
    out[base + tid]       = d0 * inv * g[tid] + bt[tid];
    out[base + tid + 256] = d1 * inv * g[tid + 256] + bt[tid + 256];
}

__global__ __launch_bounds__(256) void add2_kernel(
    const float* __restrict__ a, const float* __restrict__ b,
    float* __restrict__ o, int n)
{
    int i = blockIdx.x * 256 + threadIdx.x;
    if (i < n) o[i] = a[i] + b[i];
}

// ---------------------------------------------------------------------------
// Input stage: LN over C=16 then scale by softmax(feature_importance)
// ---------------------------------------------------------------------------
__global__ __launch_bounds__(256) void input_norm_kernel(
    const float* __restrict__ x, const float* __restrict__ fi,
    const float* __restrict__ g, const float* __restrict__ bb,
    float* __restrict__ xw)
{
    int i = blockIdx.x * 256 + threadIdx.x;
    if (i >= kBL) return;
    const float* xr = x + (size_t)i * kC;
    float m = 0.f;
    for (int c = 0; c < kC; ++c) m += xr[c];
    m *= (1.f / kC);
    float var = 0.f;
    for (int c = 0; c < kC; ++c) { float d = xr[c] - m; var += d * d; }
    var *= (1.f / kC);
    float inv = rsqrtf(var + 1e-5f);
    float fm = -3.4e38f;
    for (int c = 0; c < kC; ++c) fm = fmaxf(fm, fi[c]);
    float es[kC]; float fs = 0.f;
    for (int c = 0; c < kC; ++c) { es[c] = __expf(fi[c] - fm); fs += es[c]; }
    float finv = 1.f / fs;
    for (int c = 0; c < kC; ++c)
        xw[(size_t)i * kC + c] = ((xr[c] - m) * inv * g[c] + bb[c]) * es[c] * finv;
}

// token conv embedding (circular pad) + positional + temporal, and residual proj
__global__ __launch_bounds__(256) void embed_kernel(
    const float* __restrict__ xw, const float* __restrict__ tf,
    const float* __restrict__ tokw, const float* __restrict__ tempw,
    const float* __restrict__ tempb, const float* __restrict__ resw,
    const float* __restrict__ resb, float* __restrict__ h0,
    float* __restrict__ resid)
{
    int idx = blockIdx.x * 256 + threadIdx.x;
    if (idx >= kBL * kD) return;
    int d = idx % kD; int bl = idx / kD; int b = bl / kL; int l = bl % kL;
    int lm1 = (l == 0) ? kL - 1 : l - 1;
    int lp1 = (l == kL - 1) ? 0 : l + 1;
    const float* x0 = xw + (size_t)(b * kL + lm1) * kC;
    const float* x1 = xw + (size_t)bl * kC;
    const float* x2 = xw + (size_t)(b * kL + lp1) * kC;
    float acc = 0.f, r = resb[d];
    for (int c = 0; c < kC; ++c) {
        acc += x0[c] * tokw[(0 * kC + c) * kD + d]
             + x1[c] * tokw[(1 * kC + c) * kD + d]
             + x2[c] * tokw[(2 * kC + c) * kD + d];
        r += x1[c] * resw[c * kD + d];
    }
    int de = d & ~1;
    float div = __expf((float)de * (-9.210340372f / (float)kD));
    float ang = (float)l * div;
    acc += (d & 1) ? __cosf(ang) : __sinf(ang);
    const float* tr = tf + (size_t)bl * kT;
    for (int t = 0; t < kT; ++t) acc += tr[t] * tempw[t * kD + d];
    acc += tempb[d];
    h0[idx] = acc;
    resid[idx] = r;
}

// ---------------------------------------------------------------------------
// ProbSparse attention pieces (small FLOPs -> scalar/LDS kernels)
// ---------------------------------------------------------------------------
__global__ void vmean_kernel(const float* __restrict__ v, float* __restrict__ vm)
{
    int idx = blockIdx.x * blockDim.x + threadIdx.x;
    if (idx >= kB * kH * kDh) return;
    int b = idx / (kH * kDh); int col = idx % (kH * kDh);
    float s = 0.f;
    for (int l = 0; l < kL; ++l) s += v[(size_t)(b * kL + l) * kD + col];
    vm[idx] = s * (1.f / kL);
}

__global__ void ctx_init_kernel(const float* __restrict__ vm, float* __restrict__ ctx)
{
    int idx = blockIdx.x * blockDim.x + threadIdx.x;
    if (idx >= kBL * kD) return;
    int col = idx % kD; int bl = idx / kD; int b = bl / kL;
    ctx[idx] = vm[b * (kH * kDh) + col];
}

__global__ __launch_bounds__(256) void mscore_kernel(
    const float* __restrict__ q, const float* __restrict__ k,
    float* __restrict__ M)
{
    int idx = blockIdx.x * 256 + threadIdx.x;
    if (idx >= kB * kH * kL) return;
    int l = idx % kL; int bh = idx / kL; int b = bh / kH; int h = bh % kH;
    const float* qr = q + (size_t)(b * kL + l) * kD + h * kDh;
    float mx = -3.4e38f, sm = 0.f;
    for (int s = 0; s < kU; ++s) {
        unsigned ks = (2654435761u * (unsigned)l + 40503u * (unsigned)(s + 1) + 12345u) & (kL - 1);
        const float* kr = k + (size_t)(b * kL + (int)ks) * kD + h * kDh;
        float dot = 0.f;
        for (int d = 0; d < kDh; ++d) dot += qr[d] * kr[d];
        mx = fmaxf(mx, dot); sm += dot;
    }
    M[idx] = mx - sm * (1.f / kU);
}

__global__ __launch_bounds__(256) void topk_kernel(
    const float* __restrict__ M, int* __restrict__ topidx)
{
    __shared__ float sm[kL];
    __shared__ float sv[256];
    __shared__ int   si[256];
    int bh = blockIdx.x, tid = threadIdx.x;
    for (int j = tid; j < kL; j += 256) sm[j] = M[(size_t)bh * kL + j];
    __syncthreads();
    for (int t = 0; t < kU; ++t) {
        float best = -3.4e38f; int bi = 0;
        for (int j = tid; j < kL; j += 256)
            if (sm[j] > best) { best = sm[j]; bi = j; }
        sv[tid] = best; si[tid] = bi;
        __syncthreads();
        for (int o = 128; o > 0; o >>= 1) {
            if (tid < o && sv[tid + o] > sv[tid]) { sv[tid] = sv[tid + o]; si[tid] = si[tid + o]; }
            __syncthreads();
        }
        if (tid == 0) { topidx[bh * kU + t] = si[0]; sm[si[0]] = -3.4e38f; }
        __syncthreads();
    }
}

__global__ __launch_bounds__(256) void attn_update_kernel(
    const float* __restrict__ q, const float* __restrict__ k,
    const float* __restrict__ v, const int* __restrict__ topidx,
    float* __restrict__ ctx)
{
    __shared__ float qs[kDh];
    __shared__ float sc[kL];
    __shared__ float red[256];
    int bh = blockIdx.x / kU, ui = blockIdx.x % kU;
    int b = bh / kH, h = bh % kH;
    int tid = threadIdx.x;
    int l0 = topidx[bh * kU + ui];
    if (tid < kDh) qs[tid] = q[(size_t)(b * kL + l0) * kD + h * kDh + tid];
    __syncthreads();
    float lmax = -3.4e38f;
    for (int l = tid; l < kL; l += 256) {
        const float* kr = k + (size_t)(b * kL + l) * kD + h * kDh;
        float dot = 0.f;
        for (int d = 0; d < kDh; ++d) dot += qs[d] * kr[d];
        dot *= 0.125f;
        sc[l] = dot; lmax = fmaxf(lmax, dot);
    }
    red[tid] = lmax; __syncthreads();
    for (int o = 128; o > 0; o >>= 1) { if (tid < o) red[tid] = fmaxf(red[tid], red[tid + o]); __syncthreads(); }
    float mx = red[0]; __syncthreads();
    float lsum = 0.f;
    for (int l = tid; l < kL; l += 256) { float e = __expf(sc[l] - mx); sc[l] = e; lsum += e; }
    red[tid] = lsum; __syncthreads();
    for (int o = 128; o > 0; o >>= 1) { if (tid < o) red[tid] += red[tid + o]; __syncthreads(); }
    float inv = 1.f / red[0];
    __syncthreads();
    if (tid < kDh) {
        float acc = 0.f;
        for (int l = 0; l < kL; ++l)
            acc += sc[l] * v[(size_t)(b * kL + l) * kD + h * kDh + tid];
        ctx[(size_t)(b * kL + l0) * kD + h * kDh + tid] = acc * inv;
    }
}

// ---------------------------------------------------------------------------
// output head (tiny): relu(last @ prefc) then @ fc
// ---------------------------------------------------------------------------
__global__ void prefc_kernel(const float* __restrict__ hf, const float* __restrict__ w,
                             const float* __restrict__ bias, float* __restrict__ pre)
{
    int idx = blockIdx.x * blockDim.x + threadIdx.x;
    if (idx >= kB * 256) return;
    int b = idx / 256, j = idx % 256;
    const float* hr = hf + (size_t)(b * kL + kL - 1) * kD;
    float s = bias[j];
    for (int d = 0; d < kD; ++d) s += hr[d] * w[d * 256 + j];
    pre[idx] = fmaxf(s, 0.f);
}

__global__ void fc_kernel(const float* __restrict__ pre, const float* __restrict__ w,
                          const float* __restrict__ bias, float* __restrict__ out)
{
    int b = threadIdx.x;
    if (b >= kB) return;
    float s = bias[0];
    for (int j = 0; j < 256; ++j) s += pre[b * 256 + j] * w[j];
    out[b] = s;
}

// ===========================================================================
extern "C" void kernel_launch(void* const* d_in, const int* in_sizes, int n_in,
                              void* d_out, int out_size, void* d_ws, size_t ws_size,
                              hipStream_t stream)
{
    (void)in_sizes; (void)n_in; (void)out_size; (void)ws_size;
    const float* x     = (const float*)d_in[0];
    const float* tf    = (const float*)d_in[1];
    const float* fi    = (const float*)d_in[2];
    const float* ing   = (const float*)d_in[3];
    const float* inb   = (const float*)d_in[4];
    const float* tokw  = (const float*)d_in[5];
    const float* tempw = (const float*)d_in[6];
    const float* tempb = (const float*)d_in[7];
    const float* resw  = (const float*)d_in[8];
    const float* resb  = (const float*)d_in[9];
    const float* encg  = (const float*)d_in[10];
    const float* encb  = (const float*)d_in[11];
    const float* prefw = (const float*)d_in[12];
    const float* prefb = (const float*)d_in[13];
    const float* fcw   = (const float*)d_in[14];
    const float* fcb   = (const float*)d_in[15];
    // per layer: wq bq wk bk wv bv wo bo ff1w ff1b ff2w ff2b n1g n1b n2g n2b
    const float* LW[2][16];
    for (int i = 0; i < 2; ++i)
        for (int j = 0; j < 16; ++j)
            LW[i][j] = (const float*)d_in[16 + i * 16 + j];

    // ---- workspace carve-out ----
    char* base = (char*)d_ws;
    size_t off = 0;
    auto alloc = [&](size_t bytes) -> void* {
        off = (off + 255) & ~(size_t)255;
        void* p = base + off;
        off += bytes;
        return p;
    };
    float*  xw    = (float*)alloc((size_t)kBL * kC * 4);
    float*  h0    = (float*)alloc((size_t)kBL * kD * 4);
    float*  resid = (float*)alloc((size_t)kBL * kD * 4);
    float*  qb    = (float*)alloc((size_t)kBL * kD * 4);   // reused as t1
    float*  kb    = (float*)alloc((size_t)kBL * kD * 4);   // reused as t2
    float*  vb    = (float*)alloc((size_t)kBL * kD * 4);
    float*  ctx   = (float*)alloc((size_t)kBL * kD * 4);
    bf16_t* hb    = (bf16_t*)alloc((size_t)kBL * kD * 2);
    bf16_t* yb    = (bf16_t*)alloc((size_t)kBL * kF * 2);
    float*  Mbuf  = (float*)alloc((size_t)kB * kH * kL * 4);
    int*    tidx  = (int*)alloc((size_t)kB * kH * kU * 4);
    float*  vmn   = (float*)alloc((size_t)kB * kH * kDh * 4);
    float*  pre   = (float*)alloc((size_t)kB * 256 * 4);
    bf16_t* wp[2][6];
    for (int i = 0; i < 2; ++i) {
        wp[i][0] = (bf16_t*)alloc((size_t)kD * kD * 2);  // wq
        wp[i][1] = (bf16_t*)alloc((size_t)kD * kD * 2);  // wk
        wp[i][2] = (bf16_t*)alloc((size_t)kD * kD * 2);  // wv
        wp[i][3] = (bf16_t*)alloc((size_t)kD * kD * 2);  // wo
        wp[i][4] = (bf16_t*)alloc((size_t)kD * kF * 2);  // ff1
        wp[i][5] = (bf16_t*)alloc((size_t)kF * kD * 2);  // ff2
    }

    auto packW = [&](const float* W, bf16_t* Wp, int K, int N) {
        int n = K * N;
        pack_w_kernel<<<dim3((n + 255) / 256), dim3(256), 0, stream>>>(W, Wp, K, N);
    };
    auto gemm = [&](const bf16_t* A, const bf16_t* Bp, const float* bias,
                    float* Cf, bf16_t* Cb, int M, int N, int K, int act) {
        gemm_bf16_wmma<<<dim3(M / 128, N / 64), dim3(256), 0, stream>>>(
            A, Bp, bias, Cf, Cb, M, N, K, act);
    };
    auto tobf = [&](const float* s, bf16_t* d, int n) {
        f32_to_bf16_kernel<<<dim3((n + 255) / 256), dim3(256), 0, stream>>>(s, d, n);
    };

    // ---- weight packing (deterministic every call) ----
    for (int i = 0; i < 2; ++i) {
        packW(LW[i][0],  wp[i][0], kD, kD);
        packW(LW[i][2],  wp[i][1], kD, kD);
        packW(LW[i][4],  wp[i][2], kD, kD);
        packW(LW[i][6],  wp[i][3], kD, kD);
        packW(LW[i][8],  wp[i][4], kD, kF);
        packW(LW[i][10], wp[i][5], kF, kD);
    }

    // ---- embedding stage ----
    input_norm_kernel<<<dim3((kBL + 255) / 256), dim3(256), 0, stream>>>(x, fi, ing, inb, xw);
    embed_kernel<<<dim3((kBL * kD) / 256), dim3(256), 0, stream>>>(
        xw, tf, tokw, tempw, tempb, resw, resb, h0, resid);

    // ---- encoder layers ----
    for (int i = 0; i < 2; ++i) {
        const float* bq  = LW[i][1];  const float* bk2 = LW[i][3];
        const float* bv2 = LW[i][5];  const float* bo  = LW[i][7];
        const float* f1b = LW[i][9];  const float* f2b = LW[i][11];
        const float* n1g = LW[i][12]; const float* n1b = LW[i][13];
        const float* n2g = LW[i][14]; const float* n2b = LW[i][15];

        tobf(h0, hb, kBL * kD);
        gemm(hb, wp[i][0], bq,  qb, nullptr, kBL, kD, kD, 0);
        gemm(hb, wp[i][1], bk2, kb, nullptr, kBL, kD, kD, 0);
        gemm(hb, wp[i][2], bv2, vb, nullptr, kBL, kD, kD, 0);

        vmean_kernel<<<dim3((kB * kH * kDh + 255) / 256), dim3(256), 0, stream>>>(vb, vmn);
        ctx_init_kernel<<<dim3((kBL * kD) / 256), dim3(256), 0, stream>>>(vmn, ctx);
        mscore_kernel<<<dim3((kB * kH * kL) / 256), dim3(256), 0, stream>>>(qb, kb, Mbuf);
        topk_kernel<<<dim3(kB * kH), dim3(256), 0, stream>>>(Mbuf, tidx);
        attn_update_kernel<<<dim3(kB * kH * kU), dim3(256), 0, stream>>>(qb, kb, vb, tidx, ctx);

        tobf(ctx, hb, kBL * kD);
        gemm(hb, wp[i][3], bo, qb, nullptr, kBL, kD, kD, 0);              // attn_out -> qb
        ln512_kernel<<<dim3(kBL), dim3(256), 0, stream>>>(h0, qb, n1g, n1b, kb); // x1 -> kb
        tobf(kb, hb, kBL * kD);
        gemm(hb, wp[i][4], f1b, nullptr, yb, kBL, kF, kD, 1);             // GELU, bf16 out
        gemm(yb, wp[i][5], f2b, qb, nullptr, kBL, kD, kF, 0);             // y2 -> qb
        ln512_kernel<<<dim3(kBL), dim3(256), 0, stream>>>(kb, qb, n2g, n2b, h0); // h -> h0
    }

    // ---- final norm + residual + head ----
    ln512_kernel<<<dim3(kBL), dim3(256), 0, stream>>>(h0, nullptr, encg, encb, qb);
    add2_kernel<<<dim3((kBL * kD) / 256), dim3(256), 0, stream>>>(qb, resid, kb, kBL * kD);
    prefc_kernel<<<dim3((kB * 256 + 255) / 256), dim3(256), 0, stream>>>(kb, prefw, prefb, pre);
    fc_kernel<<<dim3(1), dim3(32), 0, stream>>>(pre, fcw, fcb, (float*)d_out);
}
